// LSTM_CRF_Model_21337397526689
// MI455X (gfx1250) — matrix operand order repaired and checked by reading, not compiled, not run on hardware
//
#include <hip/hip_runtime.h>
#include <hip/hip_bf16.h>
#include <math.h>
#include <stdint.h>

typedef __attribute__((ext_vector_type(16))) _Float16 v16h;
typedef __attribute__((ext_vector_type(8)))  _Float16 v8h;
typedef __attribute__((ext_vector_type(8)))  float    v8f;

#define Bn     128
#define Ln     256
#define En     128
#define RDn    100
#define Hn     256
#define Cn     33
#define DINP   448      // 428 padded to 14*32
#define FOURH  1024
#define CPAD   64
#define STARTT 31
#define ENDT   32

// ---------------------------------------------------------------------------
// A-operand loader for V_WMMA_F32_16X16X32_F16 (16x32 f16, row-major source).
// Per ISA 7.12.2: lanes 0-15 hold row=lane, K {k0..k0+7, k0+16..k0+23};
// lanes 16-31 hold row=lane-16, K {k0+8..k0+15, k0+24..k0+31}.
// Caller passes row = m0 + (lane&15), k = k0 + ((lane&16)?8:0).
// ---------------------------------------------------------------------------
__device__ __forceinline__ v16h load_a16(const _Float16* base, int ld, int row, int k) {
    const _Float16* p = base + (size_t)row * ld + k;
    union { v16h v; v8h h[2]; } u;
    u.h[0] = *(const v8h*)(p);
    u.h[1] = *(const v8h*)(p + 16);
    return u.v;
}

// ---------------------------------------------------------------------------
// CDNA5 async global->LDS copy (GVS mode, §15.18.3 opcode 98) + drain.
// vdst = LDS byte address, vaddr = 32-bit offset, saddr = uniform 64-bit base.
// Tracked by ASYNCcnt; drained with s_wait_asynccnt + workgroup barrier.
// ---------------------------------------------------------------------------
__device__ __forceinline__ void async_ld_b128(uint32_t lds_addr, uint32_t goff, uint64_t gbase) {
    asm volatile("global_load_async_to_lds_b128 %0, %1, %2"
                 :: "v"(lds_addr), "v"(goff), "s"(gbase) : "memory");
}
__device__ __forceinline__ void wait_async0() {
    asm volatile("s_wait_asynccnt 0x0" ::: "memory");
}

// ---------------------------------------------------------------------------
// Pack weight W [Nact, Kact] (row-major f32, used as B in y = x @ W^T) into
// WMMA B-operand tiles: tile (kt,nt) -> 32 lanes x 16 halfs contiguous, so the
// GEMM loop does one 32B load per lane per tile.
// B layout (ISA): lanes 0-15 hold col n=lane, K {0..7,16..23}; lanes 16-31 K {8..15,24..31}.
// ---------------------------------------------------------------------------
__global__ void pack_b_kernel(const float* __restrict__ W, _Float16* __restrict__ out,
                              int Kact, int Nact, int KT, int NT) {
    int idx = blockIdx.x * 256 + threadIdx.x;
    if (idx >= KT * NT * 512) return;
    int e    = idx & 15;
    int lane = (idx >> 4) & 31;
    int tile = idx >> 9;
    int nt   = tile % NT;
    int kt   = tile / NT;
    int n = nt * 16 + (lane & 15);
    int k = kt * 32 + ((lane & 16) ? 8 : 0) + (e & 7) + ((e & 8) ? 16 : 0);
    float v = (k < Kact && n < Nact) ? W[(size_t)n * Kact + k] : 0.0f;
    out[idx] = (_Float16)v;
}

// ---------------------------------------------------------------------------
// Embedding gather -> f16 activation matrix [B*L, 448] (428 real + 20 zero pad)
// ---------------------------------------------------------------------------
__global__ void gather_kernel(const int* __restrict__ data, const int* __restrict__ onerad,
                              const float* __restrict__ embed, const float* __restrict__ rad,
                              _Float16* __restrict__ emb) {
    size_t idx = (size_t)blockIdx.x * 256 + threadIdx.x;
    if (idx >= (size_t)(Bn * Ln) * DINP) return;
    int col = (int)(idx % DINP);
    int row = (int)(idx / DINP);          // row = b*L + l
    float v = 0.0f;
    if (col < En)            v = embed[(size_t)data[row] * En + col];
    else if (col < En + RDn) v = rad[(size_t)onerad[row]               * RDn + (col - En)];
    else if (col < En+2*RDn) v = rad[(size_t)onerad[Bn*Ln + row]       * RDn + (col - En - RDn)];
    else if (col < En+3*RDn) v = rad[(size_t)onerad[2*Bn*Ln + row]     * RDn + (col - En - 2*RDn)];
    emb[idx] = (_Float16)v;
}

__global__ void bias_kernel(const float* bihf, const float* bhhf,
                            const float* bihb, const float* bhhb, float* biasx) {
    int i = blockIdx.x * 256 + threadIdx.x;
    if (i < FOURH) {
        biasx[i]         = bihf[i] + bhhf[i];
        biasx[FOURH + i] = bihb[i] + bhhb[i];
    }
}

// ---------------------------------------------------------------------------
// Input projection (both directions): xg[dir][l][b][4H] = emb @ Wih^T + bias.
// Each wave: one 16-row M tile x 4 N tiles (64 cols), K = 14 tiles of 32.
// ---------------------------------------------------------------------------
__global__ __launch_bounds__(256) void xg_gemm_kernel(const _Float16* __restrict__ emb,
                                                      const _Float16* __restrict__ WihB,
                                                      const float* __restrict__ biasx,
                                                      float* __restrict__ xg) {
    int wave = threadIdx.x >> 5, lane = threadIdx.x & 31;
    int job = blockIdx.x * 8 + wave;          // 65536 wave jobs
    int dir = job >> 15;
    int rem = job & 32767;
    int mt  = rem >> 4;                       // 0..2047 (rows of B*L)
    int ns  = rem & 15;                       // strip of 64 output cols
    int lane15 = lane & 15;
    int hi8 = (lane & 16) ? 8 : 0;
    const _Float16* Bw = WihB + (size_t)dir * (14 * 64 * 512);
    v8f acc[4] = {};
    int arow = mt * 16 + lane15;
    for (int kt = 0; kt < 14; ++kt) {
        v16h a = load_a16(emb, DINP, arow, kt * 32 + hi8);
#pragma unroll
        for (int jn = 0; jn < 4; ++jn) {
            int nt = ns * 4 + jn;
            v16h b = *(const v16h*)(Bw + ((size_t)(kt * 64 + nt)) * 512 + lane * 16);
            acc[jn] = __builtin_amdgcn_wmma_f32_16x16x32_f16(false, a, false, b,
                                                             (short)0, acc[jn], false, false);
        }
    }
#pragma unroll
    for (int jn = 0; jn < 4; ++jn) {
        int n = (ns * 4 + jn) * 16 + lane15;
        float bias = biasx[dir * FOURH + n];
#pragma unroll
        for (int r = 0; r < 8; ++r) {
            int m = mt * 16 + hi8 + r;        // m = b*L + l
            int b = m >> 8, l = m & 255;
            xg[(((size_t)dir * Ln + l) * Bn + b) * FOURH + n] = acc[jn][r] + bias;
        }
    }
}

// ---------------------------------------------------------------------------
// Bi-LSTM recurrence. Batch-parallel: block = (dir, 16-row batch chunk).
// h(t) lives in LDS (16x256 f16 = 8KB), c lives in WMMA-layout VGPRs for the
// entire 256-step loop. The xg[t] pre-activation slab (64KB/block/step) is
// double-buffered in LDS and prefetched one step ahead with
// GLOBAL_LOAD_ASYNC_TO_LDS_B128, so HBM/L2 latency overlaps the WMMA chain of
// the current step instead of sitting on the 256-step serial path.
// ---------------------------------------------------------------------------
__global__ __launch_bounds__(256) void lstm_kernel(const float* __restrict__ xg,
                                                   const _Float16* __restrict__ WhhB,
                                                   _Float16* __restrict__ hcat) {
    int dir   = blockIdx.x >> 3;
    int chunk = blockIdx.x & 7;
    int rb    = chunk * 16;                   // batch rows rb..rb+15
    int tid   = threadIdx.x;
    int wave = tid >> 5, lane = tid & 31;
    int lane15 = lane & 15;
    int hi8 = (lane & 16) ? 8 : 0;
    int nh_base = wave * 2;                   // 8 waves x 2 H-tiles = all 16
    const _Float16* Bw = WhhB + (size_t)dir * (8 * 64 * 512);

    __shared__ _Float16 hbuf[16 * 256];       // h(t-1), 8KB
    __shared__ float    xgbuf[2][16 * 1024];  // staged xg slabs, 2 x 64KB

    for (int i = tid; i < 16 * 256; i += 256) hbuf[i] = (_Float16)0.0f;

    // Prologue: stage xg slab for t=0.
    {
        int l0 = dir ? (Ln - 1) : 0;
        uint64_t gb = (uint64_t)(uintptr_t)(xg + ((size_t)dir * Ln + l0) * (Bn * FOURH)
                                               + (size_t)rb * FOURH);
        uint32_t lb = (uint32_t)(uintptr_t)(&xgbuf[0][0]);
#pragma unroll
        for (int i = 0; i < 16; ++i) {
            uint32_t off = (uint32_t)(tid * 16 + i * 4096);
            async_ld_b128(lb + off, off, gb);
        }
    }
    wait_async0();
    __syncthreads();

    v8f c_st[2] = {};
    for (int t = 0; t < Ln; ++t) {
        int l   = dir ? (Ln - 1 - t) : t;
        int buf = t & 1;

        // Kick off async staging of xg for step t+1 into the other buffer.
        if (t + 1 < Ln) {
            int l2 = dir ? (Ln - 2 - t) : (t + 1);
            uint64_t gb = (uint64_t)(uintptr_t)(xg + ((size_t)dir * Ln + l2) * (Bn * FOURH)
                                                   + (size_t)rb * FOURH);
            uint32_t lb = (uint32_t)(uintptr_t)(&xgbuf[buf ^ 1][0]);
#pragma unroll
            for (int i = 0; i < 16; ++i) {
                uint32_t off = (uint32_t)(tid * 16 + i * 4096);
                async_ld_b128(lb + off, off, gb);
            }
        }

        // Accumulator preload: g = xg[t] (from staged LDS slab, WMMA C layout).
        const float* xs = &xgbuf[buf][0];
        v8f acc[2][4];
#pragma unroll
        for (int j = 0; j < 2; ++j) {
            int nh = nh_base + j;
#pragma unroll
            for (int g = 0; g < 4; ++g) {     // gate order i,f,g,o
                int n = g * Hn + nh * 16 + lane15;
                v8f a;
#pragma unroll
                for (int r = 0; r < 8; ++r) a[r] = xs[(hi8 + r) * FOURH + n];
                acc[j][g] = a;
            }
        }

        // g += h(t-1) @ Whh^T
        for (int kt = 0; kt < 8; ++kt) {
            v16h av = load_a16(hbuf, 256, lane15, kt * 32 + hi8);
#pragma unroll
            for (int j = 0; j < 2; ++j) {
                int nh = nh_base + j;
#pragma unroll
                for (int g = 0; g < 4; ++g) {
                    int nt = g * 16 + nh;
                    v16h bv = *(const v16h*)(Bw + ((size_t)(kt * 64 + nt)) * 512 + lane * 16);
                    acc[j][g] = __builtin_amdgcn_wmma_f32_16x16x32_f16(false, av, false, bv,
                                                                       (short)0, acc[j][g],
                                                                       false, false);
                }
            }
        }
        __syncthreads();                      // all waves done reading h(t-1)/xgbuf[buf]

        _Float16* hg = hcat + (size_t)l * (Bn * 2 * Hn) + (size_t)dir * Hn;
#pragma unroll
        for (int j = 0; j < 2; ++j) {
            int nh = nh_base + j;
            int col = nh * 16 + lane15;
            v8f c = c_st[j];
#pragma unroll
            for (int r = 0; r < 8; ++r) {
                float ig = 1.0f / (1.0f + expf(-acc[j][0][r]));
                float fg = 1.0f / (1.0f + expf(-acc[j][1][r]));
                float gg = tanhf(acc[j][2][r]);
                float og = 1.0f / (1.0f + expf(-acc[j][3][r]));
                float cc = fg * c[r] + ig * gg;
                c[r] = cc;
                _Float16 hv = (_Float16)(og * tanhf(cc));
                int lrow = hi8 + r;
                hbuf[lrow * 256 + col] = hv;
                hg[(size_t)(rb + lrow) * (2 * Hn) + col] = hv;
            }
            c_st[j] = c;
        }
        wait_async0();                        // own async copies for t+1 landed
        __syncthreads();                      // h(t) + staged slab visible to all
    }
}

// ---------------------------------------------------------------------------
// Emission: [L*B, 512] @ Wcls^T (+bcls) -> emis[b][l][64] (33 real cols).
// ---------------------------------------------------------------------------
__global__ __launch_bounds__(256) void emis_kernel(const _Float16* __restrict__ hcat,
                                                   const _Float16* __restrict__ WclsB,
                                                   const float* __restrict__ bcls,
                                                   float* __restrict__ emis) {
    int wave = threadIdx.x >> 5, lane = threadIdx.x & 31;
    int mt = blockIdx.x * 8 + wave;           // 0..2047, rows = l*B + b
    int lane15 = lane & 15;
    int hi8 = (lane & 16) ? 8 : 0;
    v8f acc[4] = {};
    int arow = mt * 16 + lane15;
    for (int kt = 0; kt < 16; ++kt) {
        v16h a = load_a16(hcat, 2 * Hn, arow, kt * 32 + hi8);
#pragma unroll
        for (int jn = 0; jn < 4; ++jn) {
            v16h b = *(const v16h*)(WclsB + ((size_t)(kt * 4 + jn)) * 512 + lane * 16);
            acc[jn] = __builtin_amdgcn_wmma_f32_16x16x32_f16(false, a, false, b,
                                                             (short)0, acc[jn], false, false);
        }
    }
#pragma unroll
    for (int jn = 0; jn < 4; ++jn) {
        int n = jn * 16 + lane15;
        float bias = (n < Cn) ? bcls[n] : 0.0f;
#pragma unroll
        for (int r = 0; r < 8; ++r) {
            int row = mt * 16 + hi8 + r;      // row = l*128 + b
            int l = row >> 7, b = row & 127;
            emis[((size_t)b * Ln + l) * CPAD + n] = acc[jn][r] + bias;
        }
    }
}

// ---------------------------------------------------------------------------
// CRF: golden path score + logsumexp forward scan. One block (2 waves) per
// batch element; scores double-buffered in LDS. Deterministic.
// ---------------------------------------------------------------------------
__global__ __launch_bounds__(64) void crf_kernel(const int* __restrict__ tags,
                                                 const float* __restrict__ emis,
                                                 const float* __restrict__ trans,
                                                 float* __restrict__ partial) {
    int b = blockIdx.x;
    int tid = threadIdx.x;
    __shared__ float sc[2][Cn];
    __shared__ float s_golden;
    __shared__ int   s_len;
    const float* eb = emis + (size_t)b * Ln * CPAD;
    const int*   tb = tags + b * Ln;
    if (tid == 0) {
        int len = 0;
        for (int l = 0; l < Ln; ++l) if (tb[l] != 0) len++;
        s_len = len;
        float gold = 0.0f;
        int prev = STARTT;
        for (int l = 0; l < len; ++l) {
            int cur = tb[l];
            gold += eb[l * CPAD + cur] + trans[prev * Cn + cur];
            prev = cur;
        }
        s_golden = gold;
    }
    __syncthreads();
    int len = s_len;
    if (tid < Cn) sc[0][tid] = (len > 0) ? (eb[tid] + trans[STARTT * Cn + tid]) : 0.0f;
    __syncthreads();
    int cur = 0;
    for (int t = 1; t < len; ++t) {
        int nxt = cur ^ 1;
        if (tid < Cn) {
            float m = -1e30f;
            for (int p = 0; p < Cn; ++p) m = fmaxf(m, trans[p * Cn + tid] + sc[cur][p]);
            float s = 0.0f;
            for (int p = 0; p < Cn; ++p) s += expf(trans[p * Cn + tid] + sc[cur][p] - m);
            sc[nxt][tid] = eb[t * CPAD + tid] + m + logf(s);
        }
        __syncthreads();
        cur = nxt;
    }
    if (tid == 0) partial[b] = sc[cur][ENDT] - s_golden;
}

__global__ __launch_bounds__(128) void reduce_kernel(const float* __restrict__ partial,
                                                     float* __restrict__ out) {
    __shared__ float buf[128];
    buf[threadIdx.x] = partial[threadIdx.x];
    __syncthreads();
    for (int s = 64; s > 0; s >>= 1) {
        if (threadIdx.x < s) buf[threadIdx.x] += buf[threadIdx.x + s];
        __syncthreads();
    }
    if (threadIdx.x == 0) out[0] = buf[0] / (float)Bn;
}

// ---------------------------------------------------------------------------
extern "C" void kernel_launch(void* const* d_in, const int* in_sizes, int n_in,
                              void* d_out, int out_size, void* d_ws, size_t ws_size,
                              hipStream_t stream) {
    const int*   batch_data   = (const int*)d_in[0];
    const int*   batch_onerad = (const int*)d_in[1];
    const int*   batch_tag    = (const int*)d_in[2];
    const float* embed        = (const float*)d_in[3];
    const float* rad_embed    = (const float*)d_in[4];
    const float* Wih_f = (const float*)d_in[5];
    const float* Whh_f = (const float*)d_in[6];
    const float* bih_f = (const float*)d_in[7];
    const float* bhh_f = (const float*)d_in[8];
    const float* Wih_b = (const float*)d_in[9];
    const float* Whh_b = (const float*)d_in[10];
    const float* bih_b = (const float*)d_in[11];
    const float* bhh_b = (const float*)d_in[12];
    const float* Wcls  = (const float*)d_in[13];
    const float* bcls  = (const float*)d_in[14];
    const float* trans = (const float*)d_in[15];

    char* ws = (char*)d_ws;
    _Float16* emb   = (_Float16*)(ws);                    //  29,360,128 B
    _Float16* WihB  = (_Float16*)(ws +  29360128ull);     //   1,835,008 B (2 dirs)
    _Float16* WhhB  = (_Float16*)(ws +  31195136ull);     //   1,048,576 B (2 dirs)
    _Float16* WclsB = (_Float16*)(ws +  32243712ull);     //      65,536 B
    float*    biasx = (float*)   (ws +  32309248ull);     //       8,192 B
    float*    xg    = (float*)   (ws +  32317440ull);     // 268,435,456 B
    _Float16* hcat  = (_Float16*)(ws + 300752896ull);     //  33,554,432 B
    float*    emis  = (float*)   (ws + 334307328ull);     //   8,388,608 B
    float*    part  = (float*)   (ws + 342695936ull);     //         512 B

    gather_kernel<<<57344, 256, 0, stream>>>(batch_data, batch_onerad, embed, rad_embed, emb);
    pack_b_kernel<<<1792, 256, 0, stream>>>(Wih_f, WihB,                428, FOURH, 14, 64);
    pack_b_kernel<<<1792, 256, 0, stream>>>(Wih_b, WihB + 14*64*512,    428, FOURH, 14, 64);
    pack_b_kernel<<<1024, 256, 0, stream>>>(Whh_f, WhhB,                Hn,  FOURH,  8, 64);
    pack_b_kernel<<<1024, 256, 0, stream>>>(Whh_b, WhhB + 8*64*512,     Hn,  FOURH,  8, 64);
    pack_b_kernel<<< 128, 256, 0, stream>>>(Wcls,  WclsB,               2*Hn, Cn,   16,  4);
    bias_kernel<<<4, 256, 0, stream>>>(bih_f, bhh_f, bih_b, bhh_b, biasx);

    xg_gemm_kernel<<<8192, 256, 0, stream>>>(emb, WihB, biasx, xg);
    lstm_kernel<<<16, 256, 0, stream>>>(xg, WhhB, hcat);
    emis_kernel<<<256, 256, 0, stream>>>(hcat, WclsB, bcls, emis);
    crf_kernel<<<128, 64, 0, stream>>>(batch_tag, emis, trans, part);
    reduce_kernel<<<1, 128, 0, stream>>>(part, (float*)d_out);
}
